// EdgeTypeAwareGAT_66039417143455
// MI455X (gfx1250) — matrix operand order repaired and checked by reading, not compiled, not run on hardware
//
#include <hip/hip_runtime.h>
#include <hip/hip_bf16.h>
#include <math.h>

// ---------------------------------------------------------------------------
// EdgeTypeAwareGAT for MI455X (gfx1250, wave32, WMMA).
//
// Dense GEMMs run on v_wmma_f32_16x16x32_f16 (f16 A/B, f32 accum):
//  - weights are pre-packed into WMMA B-fragment order (lane-contiguous 32B
//    chunks -> global_load_b128 pairs instead of 16 strided b16 loads)
//  - each wave computes a 16x64 output strip: 1 A fragment feeds 4 WMMAs
// Edge-indexed work (segment softmax, aggregation) is bandwidth-bound ->
// coalesced gathers + global_atomic_add_f32 scatters along the channel dim.
// Edge-feature linear folds into a 9xH dot table (eattr = onehot(type)++w).
// ---------------------------------------------------------------------------

typedef __attribute__((ext_vector_type(16))) _Float16 v16h;
typedef __attribute__((ext_vector_type(8)))  float    v8f;

#define NNODES 20000
#define NEDGES 320000
#define NEA    (NNODES + NEDGES)   // edges + self loops = 340000
#define FIN    64
#define DH     64                  // per-head channels
#define NET    8                   // num edge types
#define MHID   128
#define BNEPS  1e-5f

// ---------------- workspace layout (floats from (float*)d_ws) --------------
// packed f16 weights at the front (110592 halves = 55296 float slots)
#define W16_IN   0           // [64 x 64]
#define W16_L0   4096        // [64 x 256]
#define W16_L1   20480       // [256 x 256]
#define W16_L2   86016       // [256 x 64]
#define W16_OUT  102400      // [64 x 128]
#define OFF_HA   55296ull                      // h   [N,256] (current acts)
#define OFF_XL   (OFF_HA  + 5120000ull)        // xl  [N,256]
#define OFF_OUT  (OFF_XL  + 5120000ull)        // agg [N,256]
#define OFF_ASRC (OFF_OUT + 5120000ull)        // [N,4]
#define OFF_ADST (OFF_ASRC + 80000ull)         // [N,4]
#define OFF_AMAX (OFF_ADST + 80000ull)         // [N,4]
#define OFF_DEN  (OFF_AMAX + 80000ull)         // [N,4]
#define OFF_ALPH (OFF_DEN  + 80000ull)         // [Ea,4]
#define OFF_TCNT (OFF_ALPH + 1360000ull)       // [N,8] per-type in-counts
#define OFF_WSUM (OFF_TCNT + 160000ull)        // [N]
#define OFF_CNT  (OFF_WSUM + 20000ull)         // [N]
#define OFF_TD   (OFF_CNT  + 20000ull)         // [9,4] dot table (pad 64)
#define OFF_MU   (OFF_TD   + 64ull)            // [256]
#define OFF_VAR  (OFF_MU   + 256ull)           // [256]
// total ~= 69.2 MB

// d_out layout: emb [N*128] | att [Ea*1] | importance [N]
#define EMB_OFF 0ull
#define ATT_OFF 2560000ull
#define IMP_OFF 2900000ull

// ------------------------------ helpers ------------------------------------
__device__ __forceinline__ float lrelu02(float x) { return x > 0.f ? x : 0.2f * x; }

// order-preserving float atomic max via int/uint monotonic mapping
__device__ __forceinline__ void atomicMaxF(float* a, float v) {
  if (v >= 0.f) atomicMax((int*)a, __float_as_int(v));
  else          atomicMin((unsigned int*)a, __float_as_uint(v));
}

__global__ void k_fill(float* __restrict__ p, float v, int n) {
  int i = blockIdx.x * blockDim.x + threadIdx.x;
  if (i < n) p[i] = v;
}

// ---- pack f32 weight [K,Nc] into WMMA B-fragment order (f16) --------------
// fragment (nt,kc) = 32 lanes x 16 halfs, lane-contiguous:
//   P[((nt*(K/32)+kc)*32 + lane)*16 + j] = (f16) W[(kc*32 + 2*(j&8) + (lane&16)/2
//                                                  + (j&7)) * Nc + nt*16 + (lane&15)]
__global__ void k_pack_b(const float* __restrict__ W, _Float16* __restrict__ P,
                         int K, int Nc) {
  int t = blockIdx.x * blockDim.x + threadIdx.x;
  if (t >= K * Nc) return;
  int j    = t & 15;
  int lane = (t >> 4) & 31;
  int rest = t >> 9;
  int kch  = K >> 5;
  int kc   = rest % kch;
  int nt   = rest / kch;
  int n    = nt * 16 + (lane & 15);
  int k    = kc * 32 + ((j & 8) ? 16 : 0) + ((lane & 16) ? 8 : 0) + (j & 7);
  P[t] = (_Float16)W[(size_t)k * Nc + n];
}

// -------------------- WMMA GEMM: C[M,Nc] = A[M,K] @ Bpacked (+bias) --------
// One wave per 16x64 output strip: one A fragment (float4 loads, f32->f16)
// feeds 4 v_wmma_f32_16x16x32_f16 per 32-wide K chunk.
__global__ void __launch_bounds__(32)
k_gemm_wmma(const float* __restrict__ A, const _Float16* __restrict__ Bp,
            float* __restrict__ C, const float* __restrict__ bias,
            int M, int K, int Nc) {
  const int ng   = Nc >> 6;                 // 64-col groups
  const int mt   = blockIdx.x / ng;
  const int g    = blockIdx.x % ng;
  const int lane = threadIdx.x & 31;
  const int lh   = lane >> 4;
  const int l16  = lane & 15;
  const int kch  = K >> 5;

  v8f acc0 = {}, acc1 = {}, acc2 = {}, acc3 = {};
  // A fragment rows: lanes 0-15 carry K={0..7,16..23}, lanes 16-31 K={8..15,24..31}
  const float* abase = A + (size_t)(mt * 16 + l16) * K + (lh ? 8 : 0);
  const v16h*  bbase = (const v16h*)Bp + ((size_t)(4 * g) * kch) * 32 + lane;
  const size_t bstr  = (size_t)kch * 32;    // v16h stride between n-tiles

  for (int kc = 0; kc < kch; ++kc) {
    const float4* ap = (const float4*)(abase + kc * 32);
    if (kc + 1 < kch) __builtin_prefetch(abase + (kc + 1) * 32, 0, 0);
    float4 x0 = ap[0], x1 = ap[1], x2 = ap[4], x3 = ap[5];
    v16h a;
    a[0]  = (_Float16)x0.x; a[1]  = (_Float16)x0.y;
    a[2]  = (_Float16)x0.z; a[3]  = (_Float16)x0.w;
    a[4]  = (_Float16)x1.x; a[5]  = (_Float16)x1.y;
    a[6]  = (_Float16)x1.z; a[7]  = (_Float16)x1.w;
    a[8]  = (_Float16)x2.x; a[9]  = (_Float16)x2.y;
    a[10] = (_Float16)x2.z; a[11] = (_Float16)x2.w;
    a[12] = (_Float16)x3.x; a[13] = (_Float16)x3.y;
    a[14] = (_Float16)x3.z; a[15] = (_Float16)x3.w;

    const v16h* bp = bbase + (size_t)kc * 32;
    v16h b0 = bp[0];
    v16h b1 = bp[bstr];
    v16h b2 = bp[2 * bstr];
    v16h b3 = bp[3 * bstr];

    acc0 = __builtin_amdgcn_wmma_f32_16x16x32_f16(false, a, false, b0, (short)0, acc0, false, false);
    acc1 = __builtin_amdgcn_wmma_f32_16x16x32_f16(false, a, false, b1, (short)0, acc1, false, false);
    acc2 = __builtin_amdgcn_wmma_f32_16x16x32_f16(false, a, false, b2, (short)0, acc2, false, false);
    acc3 = __builtin_amdgcn_wmma_f32_16x16x32_f16(false, a, false, b3, (short)0, acc3, false, false);
  }

  v8f accs[4] = {acc0, acc1, acc2, acc3};
#pragma unroll
  for (int sub = 0; sub < 4; ++sub) {
    int n    = (4 * g + sub) * 16 + l16;
    float bv = bias ? bias[n] : 0.f;
    float* crow = C + (size_t)(mt * 16 + lh * 8) * Nc + n;
#pragma unroll
    for (int r = 0; r < 8; ++r) crow[(size_t)r * Nc] = accs[sub][r] + bv;
  }
}

// ----------------- per-node incoming-edge stats (for self-loop attr) -------
__global__ void k_edge_stats(const int* __restrict__ ei, const int* __restrict__ et,
                             const float* __restrict__ ew,
                             float* __restrict__ tcnt, float* __restrict__ wsum,
                             float* __restrict__ cnt) {
  int e = blockIdx.x * blockDim.x + threadIdx.x;
  if (e >= NEDGES) return;
  int d = ei[NEDGES + e];
  atomicAdd(&tcnt[d * NET + et[e]], 1.f);
  atomicAdd(&wsum[d], ew[e]);
  atomicAdd(&cnt[d], 1.f);
}

// a_src[n,h] = <xl[n,h,:], att_src[h,:]>, same for a_dst
__global__ void k_att_dots(const float* __restrict__ xl, const float* __restrict__ asw,
                           const float* __restrict__ adw, float* __restrict__ a_src,
                           float* __restrict__ a_dst, int H) {
  int t = blockIdx.x * blockDim.x + threadIdx.x;
  if (t >= NNODES * H) return;
  int n = t / H, hh = t % H;
  const float* row = xl + (size_t)n * H * DH + hh * DH;
  const float* s   = asw + hh * DH;
  const float* d   = adw + hh * DH;
  float ss = 0.f, sd = 0.f;
  for (int c = 0; c < DH; ++c) { float v = row[c]; ss += v * s[c]; sd += v * d[c]; }
  a_src[t] = ss; a_dst[t] = sd;
}

// td[t,h] = <lin_edge_w[t, h*64:(h+1)*64], att_edge[h,:]>, t in 0..8 (8=weight row)
__global__ void k_td(const float* __restrict__ lew, const float* __restrict__ aew,
                     float* __restrict__ td, int H) {
  int t = threadIdx.x;
  if (t >= (NET + 1) * H) return;
  int ti = t / H, hh = t % H;
  const float* r = lew + (size_t)ti * H * DH + hh * DH;
  const float* a = aew + hh * DH;
  float s = 0.f;
  for (int c = 0; c < DH; ++c) s += r[c] * a[c];
  td[t] = s;
}

// pass 1: alpha = lrelu(a_src[src]+a_dst[dst]+a_edge); segment max -> amax[dst]
__global__ void k_alpha_max(const int* __restrict__ ei, const int* __restrict__ et,
                            const float* __restrict__ ew, const float* __restrict__ a_src,
                            const float* __restrict__ a_dst, const float* __restrict__ td,
                            const float* __restrict__ tcnt, const float* __restrict__ wsum,
                            const float* __restrict__ cnt, float* __restrict__ alpha,
                            float* __restrict__ amax, int H) {
  int t = blockIdx.x * blockDim.x + threadIdx.x;
  if (t >= NEA * H) return;
  int e = t / H, hh = t % H;
  int s, d; float ae;
  if (e < NEDGES) {
    s = ei[e]; d = ei[NEDGES + e];
    ae = td[et[e] * H + hh] + ew[e] * td[NET * H + hh];
  } else {                                   // self loop, edge_attr = mean incoming
    s = e - NEDGES; d = s;
    float c = fmaxf(cnt[s], 1.f);
    float acc = wsum[s] * td[NET * H + hh];
    for (int ti = 0; ti < NET; ++ti) acc += tcnt[s * NET + ti] * td[ti * H + hh];
    ae = acc / c;
  }
  float al = lrelu02(a_src[s * H + hh] + a_dst[d * H + hh] + ae);
  alpha[t] = al;
  atomicMaxF(&amax[d * H + hh], al);
}

// pass 2: ex = exp(alpha - amax[dst]); segment sum -> denom[dst]
__global__ void k_exp_sum(const int* __restrict__ ei, const float* __restrict__ amax,
                          float* __restrict__ alpha, float* __restrict__ denom, int H) {
  int t = blockIdx.x * blockDim.x + threadIdx.x;
  if (t >= NEA * H) return;
  int e = t / H, hh = t % H;
  int d = (e < NEDGES) ? ei[NEDGES + e] : (e - NEDGES);
  float ex = __expf(alpha[t] - amax[d * H + hh]);
  alpha[t] = ex;
  atomicAdd(&denom[d * H + hh], ex);
}

// pass 3: out[dst] += (ex/denom[dst]) * xl[src]  (coalesced along channel c)
__global__ void k_aggregate(const int* __restrict__ ei, const float* __restrict__ xl,
                            const float* __restrict__ ex, const float* __restrict__ denom,
                            float* __restrict__ out, int H) {
  int t = blockIdx.x * blockDim.x + threadIdx.x;
  if (t >= NEA * DH) return;
  int e = t / DH, c = t % DH;
  int s, d;
  if (e < NEDGES) { s = ei[e]; d = ei[NEDGES + e]; } else { s = e - NEDGES; d = s; }
  for (int hh = 0; hh < H; ++hh) {
    float att = ex[e * H + hh] / denom[d * H + hh];
    atomicAdd(&out[(size_t)d * H * DH + hh * DH + c],
              att * xl[(size_t)s * H * DH + hh * DH + c]);
  }
}

// final layer (H=1): emit attention coefficients to output
__global__ void k_att_out(const int* __restrict__ ei, const float* __restrict__ ex,
                          const float* __restrict__ denom, float* __restrict__ att) {
  int e = blockIdx.x * blockDim.x + threadIdx.x;
  if (e >= NEA) return;
  int d = (e < NEDGES) ? ei[NEDGES + e] : (e - NEDGES);
  att[e] = ex[e] / denom[d];
}

// batchnorm stats: one block per channel
__global__ void __launch_bounds__(256)
k_bn_stats(const float* __restrict__ h, float* __restrict__ mu,
           float* __restrict__ var, int odim) {
  __shared__ float ssum[256], ssq[256];
  int ch = blockIdx.x;
  float s = 0.f, q = 0.f;
  for (int n = threadIdx.x; n < NNODES; n += blockDim.x) {
    float v = h[(size_t)n * odim + ch];
    s += v; q += v * v;
  }
  ssum[threadIdx.x] = s; ssq[threadIdx.x] = q;
  __syncthreads();
  for (int o = 128; o > 0; o >>= 1) {
    if (threadIdx.x < o) { ssum[threadIdx.x] += ssum[threadIdx.x + o];
                           ssq[threadIdx.x]  += ssq[threadIdx.x + o]; }
    __syncthreads();
  }
  if (threadIdx.x == 0) {
    float m = ssum[0] / (float)NNODES;
    mu[ch]  = m;
    var[ch] = ssq[0] / (float)NNODES - m * m;
  }
}

// (GAT bias cancels through BN mean-subtraction, so it is omitted.)
__global__ void k_bn_apply(const float* __restrict__ in, const float* __restrict__ mu,
                           const float* __restrict__ var, const float* __restrict__ g,
                           const float* __restrict__ b, float* __restrict__ out,
                           int odim, int do_elu) {
  int t = blockIdx.x * blockDim.x + threadIdx.x;
  if (t >= NNODES * odim) return;
  int ch = t % odim;
  float v = (in[t] - mu[ch]) * rsqrtf(var[ch] + BNEPS) * g[ch] + b[ch];
  if (do_elu) v = v > 0.f ? v : __expf(v) - 1.f;
  out[t] = v;
}

__global__ void k_importance(const float* __restrict__ emb, const float* __restrict__ iw,
                             const float* __restrict__ ib, float* __restrict__ outp) {
  int n = blockIdx.x * blockDim.x + threadIdx.x;
  if (n >= NNODES) return;
  const float* r = emb + (size_t)n * MHID;
  float s = ib[0];
  for (int c = 0; c < MHID; ++c) s += r[c] * iw[c];
  outp[n] = 1.f / (1.f + __expf(-s));
}

// ------------------------------ launcher -----------------------------------
static inline int cdiv(int a, int b) { return (a + b - 1) / b; }

extern "C" void kernel_launch(void* const* d_in, const int* in_sizes, int n_in,
                              void* d_out, int out_size, void* d_ws, size_t ws_size,
                              hipStream_t stream) {
  // Input order = setup_inputs() dict order, params flattened depth-first in
  // declaration order: x, edge_index, edge_type, edge_weight, in_w, in_b,
  // out_w, out_b, imp_w, imp_b, then per layer {lin_w, lin_edge_w, att_src,
  // att_dst, att_edge, bias, bn_g, bn_b}.
  const float* x     = (const float*)d_in[0];
  const int*   ei    = (const int*)  d_in[1];   // [2,E] row-major: src then dst
  const int*   et    = (const int*)  d_in[2];
  const float* ew    = (const float*)d_in[3];
  const float* in_w  = (const float*)d_in[4];
  const float* in_b  = (const float*)d_in[5];
  const float* out_w = (const float*)d_in[6];
  const float* out_b = (const float*)d_in[7];
  const float* imp_w = (const float*)d_in[8];
  const float* imp_b = (const float*)d_in[9];

  struct Lp { const float *lin_w, *lew, *asrc, *adst, *aedge, *bn_g, *bn_b; };
  Lp L[3];
  for (int i = 0; i < 3; ++i) {
    const int base = 10 + 8 * i;
    L[i].lin_w = (const float*)d_in[base + 0];
    L[i].lew   = (const float*)d_in[base + 1];
    L[i].asrc  = (const float*)d_in[base + 2];
    L[i].adst  = (const float*)d_in[base + 3];
    L[i].aedge = (const float*)d_in[base + 4];
    L[i].bn_g  = (const float*)d_in[base + 6];
    L[i].bn_b  = (const float*)d_in[base + 7];
  }
  const int HEADS[3] = {4, 4, 1};
  const int CIN[3]   = {64, 256, 256};

  float*    wsf = (float*)d_ws;
  _Float16* w16 = (_Float16*)d_ws;
  float* hA   = wsf + OFF_HA;
  float* xl   = wsf + OFF_XL;
  float* agg  = wsf + OFF_OUT;
  float* aS   = wsf + OFF_ASRC;
  float* aD   = wsf + OFF_ADST;
  float* amax = wsf + OFF_AMAX;
  float* den  = wsf + OFF_DEN;
  float* alph = wsf + OFF_ALPH;
  float* tcnt = wsf + OFF_TCNT;
  float* wsum = wsf + OFF_WSUM;
  float* cnt  = wsf + OFF_CNT;
  float* td   = wsf + OFF_TD;
  float* mu   = wsf + OFF_MU;
  float* var  = wsf + OFF_VAR;

  float* emb  = (float*)d_out + EMB_OFF;
  float* attO = (float*)d_out + ATT_OFF;
  float* impO = (float*)d_out + IMP_OFF;

  const int TB = 256;

  // weights -> packed f16 B-fragments (every call; cheap and deterministic)
  k_pack_b<<<cdiv(4096,  TB), TB, 0, stream>>>(in_w,       w16 + W16_IN,  64,  64);
  k_pack_b<<<cdiv(16384, TB), TB, 0, stream>>>(L[0].lin_w, w16 + W16_L0,  64, 256);
  k_pack_b<<<cdiv(65536, TB), TB, 0, stream>>>(L[1].lin_w, w16 + W16_L1, 256, 256);
  k_pack_b<<<cdiv(16384, TB), TB, 0, stream>>>(L[2].lin_w, w16 + W16_L2, 256,  64);
  k_pack_b<<<cdiv(8192,  TB), TB, 0, stream>>>(out_w,      w16 + W16_OUT, 64, 128);

  // per-node incoming edge stats (self-loop mean attr)
  k_fill<<<cdiv(NNODES * NET, TB), TB, 0, stream>>>(tcnt, 0.f, NNODES * NET);
  k_fill<<<cdiv(NNODES, TB), TB, 0, stream>>>(wsum, 0.f, NNODES);
  k_fill<<<cdiv(NNODES, TB), TB, 0, stream>>>(cnt,  0.f, NNODES);
  k_edge_stats<<<cdiv(NEDGES, TB), TB, 0, stream>>>(ei, et, ew, tcnt, wsum, cnt);

  // input projection: hA = x @ in_w + in_b   [N,64]
  k_gemm_wmma<<<(NNODES / 16) * (FIN / 64), 32, 0, stream>>>(
      x, w16 + W16_IN, hA, in_b, NNODES, FIN, FIN);

  const int W16L[3] = {W16_L0, W16_L1, W16_L2};
  for (int l = 0; l < 3; ++l) {
    const int H = HEADS[l], K = CIN[l], HD = H * DH, odim = HD; // H=1: mean == identity
    // xl = h @ lin_w   [N, H*64]
    k_gemm_wmma<<<(NNODES / 16) * (HD / 64), 32, 0, stream>>>(
        hA, w16 + W16L[l], xl, nullptr, NNODES, K, HD);
    // attention dot products
    k_att_dots<<<cdiv(NNODES * H, TB), TB, 0, stream>>>(xl, L[l].asrc, L[l].adst, aS, aD, H);
    k_td<<<1, 64, 0, stream>>>(L[l].lew, L[l].aedge, td, H);
    // segment softmax init
    k_fill<<<cdiv(NNODES * H, TB), TB, 0, stream>>>(amax, -INFINITY, NNODES * H);
    k_fill<<<cdiv(NNODES * H, TB), TB, 0, stream>>>(den, 0.f, NNODES * H);
    k_fill<<<cdiv(NNODES * HD, TB), TB, 0, stream>>>(agg, 0.f, NNODES * HD);
    // softmax passes + aggregation
    k_alpha_max<<<cdiv(NEA * H, TB), TB, 0, stream>>>(ei, et, ew, aS, aD, td,
                                                      tcnt, wsum, cnt, alph, amax, H);
    k_exp_sum<<<cdiv(NEA * H, TB), TB, 0, stream>>>(ei, amax, alph, den, H);
    if (l == 2) k_att_out<<<cdiv(NEA, TB), TB, 0, stream>>>(ei, alph, den, attO);
    k_aggregate<<<cdiv(NEA * DH, TB), TB, 0, stream>>>(ei, xl, alph, den, agg, H);
    // batchnorm (+ELU on layers 0,1) -> next h
    k_bn_stats<<<odim, 256, 0, stream>>>(agg, mu, var, odim);
    k_bn_apply<<<cdiv(NNODES * odim, TB), TB, 0, stream>>>(
        agg, mu, var, L[l].bn_g, L[l].bn_b, hA, odim, l < 2 ? 1 : 0);
  }

  // emb = h @ out_w + out_b   [N,128]  (written straight to d_out)
  k_gemm_wmma<<<(NNODES / 16) * (MHID / 64), 32, 0, stream>>>(
      hA, w16 + W16_OUT, emb, out_b, NNODES, DH, MHID);
  // importance = sigmoid(emb @ imp_w + imp_b)
  k_importance<<<cdiv(NNODES, TB), TB, 0, stream>>>(emb, imp_w, imp_b, impO);

  (void)in_sizes; (void)n_in; (void)out_size; (void)ws_size;
}
